// critic_adq_49409303773229
// MI455X (gfx1250) — matrix operand order repaired and checked by reading, not compiled, not run on hardware
//
#include <hip/hip_runtime.h>
#include <hip/hip_bf16.h>

typedef __bf16 bf16;
typedef __attribute__((ext_vector_type(16))) __bf16 v16bf;
typedef __attribute__((ext_vector_type(4)))  __bf16 v4bf;
typedef __attribute__((ext_vector_type(8)))  float  v8f;

#define NAG    8
#define PEROBS 64
#define PERACT 32
#define FEAT   96   // PEROBS + PERACT
#define H4     64
#define H8     32
#define NUNITS 256
#define MROWS  64   // batch rows per block (4 M-tiles)

// padded LDS leading dims (bf16 elems): +8 => odd-multiple-of-4 DWORD strides,
// conflict-free ds_load_b128 A-fragments (each bank hit exactly twice)
#define LDX   104   // x tile   (96 + 8)
#define LDH   72    // h        (64 + 8)
#define LDQ   40    // q/v_self (32 + 8)
#define LDKV  232   // k/v      (224 + 8)
#define LDT   264   // t1/t2    (256 + 8)
#define LDXC  72    // xc       (64 + 8)

// ---- WMMA helpers (CDNA5 v_wmma_f32_16x16x32_bf16) -------------------------

__device__ __forceinline__ v8f wmma_bf16(v16bf a, v16bf b, v8f c) {
  return __builtin_amdgcn_wmma_f32_16x16x32_bf16(false, a, false, b, (short)0, c,
                                                 false, false);
}

// A fragment (16x32 bf16, MxK) from LDS, row-major, leading dim ld.
__device__ __forceinline__ v16bf load_A_lds(const bf16* base, int ld, int kb, int lane) {
  const bf16* p = base + (lane & 15) * ld + kb + ((lane >> 4) & 1) * 8;
  v16bf a;
#pragma unroll
  for (int e = 0; e < 8; ++e) { a[e] = p[e]; a[e + 8] = p[e + 16]; }
  return a;
}

// B fragment (32x16 bf16, KxN), B = W^T with W row-major (out x in, ld = in).
__device__ __forceinline__ v16bf load_B_w(const float* __restrict__ W, int ld,
                                          int nb, int kb, int lane) {
  const float* p = W + (size_t)(nb + (lane & 15)) * ld + kb + ((lane >> 4) & 1) * 8;
  v16bf b;
#pragma unroll
  for (int e = 0; e < 8; ++e) { b[e] = (bf16)p[e]; b[e + 8] = (bf16)p[e + 16]; }
  return b;
}

__device__ __forceinline__ v8f bias_acc(float bval) {
  v8f acc;
#pragma unroll
  for (int i = 0; i < 8; ++i) acc[i] = bval;
  return acc;
}

// C layout: lane l, element i -> row = i + 8*(l>=16), col = (l&15) (+ nt base)
__device__ __forceinline__ void store_tile_lrelu_bf16(v8f acc, bf16* dst, int ld,
                                                      int col, int lane) {
  const int rb = (lane >> 4) * 8;
#pragma unroll
  for (int i = 0; i < 8; ++i) {
    float y = fmaxf(acc[i], 0.f) + 0.01f * fminf(acc[i], 0.f);
    dst[(rb + i) * ld + col] = (bf16)y;
  }
}

__device__ __forceinline__ void store_tile_bf16(v8f acc, bf16* dst, int ld,
                                                int col, int lane) {
  const int rb = (lane >> 4) * 8;
#pragma unroll
  for (int i = 0; i < 8; ++i) dst[(rb + i) * ld + col] = (bf16)acc[i];
}

// ---- block-cooperative MLP branch: xc(64x64) ->256 lrelu ->256 lrelu ->1 ----
__device__ __forceinline__ void mlp_branch(const bf16* xc, bf16* t1, bf16* t2,
    const float* __restrict__ Wl1, const float* __restrict__ bl1,
    const float* __restrict__ Wl2, const float* __restrict__ bl2,
    const float* __restrict__ Wo,  const float* __restrict__ bo,
    float* __restrict__ outp, int tid)
{
  const int lane = tid & 31, wave = tid >> 5;

  // layer 1: 64 -> 256 (wave handles nt = 2*wave + {0,1})
#pragma unroll
  for (int i = 0; i < 2; ++i) {
    const int ntb = (wave * 2 + i) * 16;
    const v16bf B0 = load_B_w(Wl1, H4, ntb, 0, lane);
    const v16bf B1 = load_B_w(Wl1, H4, ntb, 32, lane);
    const int col = ntb + (lane & 15);
    const float bval = bl1[col];
#pragma unroll
    for (int m = 0; m < 4; ++m) {
      const bf16* A = xc + m * 16 * LDXC;
      v8f acc = bias_acc(bval);
      acc = wmma_bf16(load_A_lds(A, LDXC, 0, lane), B0, acc);
      acc = wmma_bf16(load_A_lds(A, LDXC, 32, lane), B1, acc);
      store_tile_lrelu_bf16(acc, t1 + m * 16 * LDT, LDT, col, lane);
    }
  }
  __syncthreads();

  // layer 2: 256 -> 256 (8 K-fragments of W held in registers per nt)
#pragma unroll 1
  for (int i = 0; i < 2; ++i) {
    const int ntb = (wave * 2 + i) * 16;
    v16bf Bf[8];
#pragma unroll
    for (int kb = 0; kb < 8; ++kb) Bf[kb] = load_B_w(Wl2, NUNITS, ntb, kb * 32, lane);
    const int col = ntb + (lane & 15);
    const float bval = bl2[col];
#pragma unroll 1
    for (int m = 0; m < 4; ++m) {
      const bf16* A = t1 + m * 16 * LDT;
      v8f acc = bias_acc(bval);
#pragma unroll
      for (int kb = 0; kb < 8; ++kb)
        acc = wmma_bf16(load_A_lds(A, LDT, kb * 32, lane), Bf[kb], acc);
      store_tile_lrelu_bf16(acc, t2 + m * 16 * LDT, LDT, col, lane);
    }
  }
  __syncthreads();

  // output mat-vec: 4 threads per row, shuffle reduce
  {
    const int r = tid >> 2, q = tid & 3;
    const bf16*  tr = t2 + r * LDT + q * 64;
    const float* wr = Wo + q * 64;
    float s = 0.f;
#pragma unroll
    for (int k = 0; k < 64; ++k) s += (float)tr[k] * wr[k];
    s += __shfl_xor(s, 1, 32);
    s += __shfl_xor(s, 2, 32);
    if (q == 0) outp[r] = s + bo[0];
  }
  __syncthreads();  // t1/t2 reuse by next branch
}

// ---- main fused kernel: 256 threads (8 waves) per 64 batch rows ------------

__global__ __launch_bounds__(256) void critic_adq_kernel(
    const float* __restrict__ obs, const float* __restrict__ act,
    const float* __restrict__ Ws,  const float* __restrict__ bs,
    const float* __restrict__ Wq,  const float* __restrict__ bq,
    const float* __restrict__ Wk,  const float* __restrict__ bk,
    const float* __restrict__ Wv,  const float* __restrict__ bv,
    const float* __restrict__ Wa,  const float* __restrict__ ba,
    const float* __restrict__ W01, const float* __restrict__ b01,
    const float* __restrict__ W02, const float* __restrict__ b02,
    const float* __restrict__ W0,  const float* __restrict__ b0,
    const float* __restrict__ W11, const float* __restrict__ b11,
    const float* __restrict__ W12, const float* __restrict__ b12,
    const float* __restrict__ W1,  const float* __restrict__ b1,
    float* __restrict__ out, int Btot)
{
  extern __shared__ char smem[];
  const int tid  = threadIdx.x;
  const int lane = tid & 31;
  const int wave = tid >> 5;
  const int b0row = blockIdx.x * MROWS;

  // LDS regions (140KB peak -> 2 blocks/WGP of the 320KB pool):
  bf16* hb   = (bf16*)smem;                    // [8][64][LDH]  73728 B
  bf16* xbuf = (bf16*)(smem + 73728);          // [64][LDX]     13312 B (per-agent)
  bf16* qb   = (bf16*)(smem + 73728);          // [64][LDQ]  (overwrites xbuf)
  bf16* vsb  = qb  + MROWS * LDQ;              // [64][LDQ]
  bf16* kbf  = vsb + MROWS * LDQ;              // [64][LDKV]
  bf16* vbf  = kbf + MROWS * LDKV;             // [64][LDKV]   (ends at 143360)
  bf16* xc   = (bf16*)smem;                    // [64][LDXC]  9216 B (overwrites hb)
  bf16* t1   = (bf16*)(smem + 9216);           // [64][LDT]  33792 B
  bf16* t2   = (bf16*)(smem + 43008);          // [64][LDT]  33792 B (ends 76800)

  // ---- stage 1: per agent, stage x then h = lrelu(x @ Ws^T + bs) ----------
  // 8 waves = 4 N-tiles x 2 M-halves; Ws fragments in registers across agents
  {
    const int ntb = (wave & 3) * 16;
    const int mh  = (wave >> 2) * 2;
    const v16bf B0 = load_B_w(Ws, FEAT, ntb, 0, lane);
    const v16bf B1 = load_B_w(Ws, FEAT, ntb, 32, lane);
    const v16bf B2 = load_B_w(Ws, FEAT, ntb, 64, lane);
    const int col = ntb + (lane & 15);
    const float bval = bs[col];
#pragma unroll 1
    for (int ag = 0; ag < NAG; ++ag) {
      // coalesced f32 -> bf16 staging of this agent's x slab (64 x 96)
      for (int u = tid; u < MROWS * 16; u += 256) {        // obs: 16 float4/row
        const int r = u >> 4, c4 = u & 15;
        float4 tv = ((const float4*)(obs + (size_t)(b0row + r) * (NAG * PEROBS)
                                         + ag * PEROBS))[c4];
        v4bf o; o[0]=(bf16)tv.x; o[1]=(bf16)tv.y; o[2]=(bf16)tv.z; o[3]=(bf16)tv.w;
        *(v4bf*)(xbuf + r * LDX + c4 * 4) = o;
      }
      for (int u = tid; u < MROWS * 8; u += 256) {         // act: 8 float4/row
        const int r = u >> 3, c4 = u & 7;
        float4 tv = ((const float4*)(act + (size_t)(b0row + r) * (NAG * PERACT)
                                         + ag * PERACT))[c4];
        v4bf o; o[0]=(bf16)tv.x; o[1]=(bf16)tv.y; o[2]=(bf16)tv.z; o[3]=(bf16)tv.w;
        *(v4bf*)(xbuf + r * LDX + PEROBS + c4 * 4) = o;
      }
      __syncthreads();
#pragma unroll
      for (int mm = 0; mm < 2; ++mm) {
        const int m = mh + mm;
        const bf16* A = xbuf + m * 16 * LDX;
        v8f acc = bias_acc(bval);
        acc = wmma_bf16(load_A_lds(A, LDX, 0, lane), B0, acc);
        acc = wmma_bf16(load_A_lds(A, LDX, 32, lane), B1, acc);
        acc = wmma_bf16(load_A_lds(A, LDX, 64, lane), B2, acc);
        store_tile_lrelu_bf16(acc, hb + (ag * MROWS + m * 16) * LDH, LDH, col, lane);
      }
      __syncthreads();   // xbuf reused by next agent
    }
  }

  // ---- stage 2: q/k/v projections; 8 waves = 2 N-tiles x 4 M-tiles --------
  {
    const int ntb = (wave & 1) * 16;
    const int m   = wave >> 1;
    const int col = ntb + (lane & 15);
    const bf16* H0 = hb + m * 16 * LDH;                 // agent 0, this M-tile
    {  // q = h_self @ Wq^T + bq
      const v16bf B0 = load_B_w(Wq, H4, ntb, 0, lane);
      const v16bf B1 = load_B_w(Wq, H4, ntb, 32, lane);
      v8f acc = bias_acc(bq[col]);
      acc = wmma_bf16(load_A_lds(H0, LDH, 0, lane), B0, acc);
      acc = wmma_bf16(load_A_lds(H0, LDH, 32, lane), B1, acc);
      store_tile_bf16(acc, qb + m * 16 * LDQ, LDQ, col, lane);
    }
    {  // v for all agents (agent 0 -> v_self)
      const v16bf B0 = load_B_w(Wv, H4, ntb, 0, lane);
      const v16bf B1 = load_B_w(Wv, H4, ntb, 32, lane);
      const float bval = bv[col];
#pragma unroll 1
      for (int ag = 0; ag < NAG; ++ag) {
        const bf16* A = hb + (ag * MROWS + m * 16) * LDH;
        v8f acc = bias_acc(bval);
        acc = wmma_bf16(load_A_lds(A, LDH, 0, lane), B0, acc);
        acc = wmma_bf16(load_A_lds(A, LDH, 32, lane), B1, acc);
        if (ag == 0) store_tile_bf16(acc, vsb + m * 16 * LDQ, LDQ, col, lane);
        else         store_tile_bf16(acc, vbf + m * 16 * LDKV + (ag - 1) * 32, LDKV, col, lane);
      }
    }
    {  // k for agents 1..7
      const v16bf B0 = load_B_w(Wk, H4, ntb, 0, lane);
      const v16bf B1 = load_B_w(Wk, H4, ntb, 32, lane);
      const float bval = bk[col];
#pragma unroll 1
      for (int ag = 1; ag < NAG; ++ag) {
        const bf16* A = hb + (ag * MROWS + m * 16) * LDH;
        v8f acc = bias_acc(bval);
        acc = wmma_bf16(load_A_lds(A, LDH, 0, lane), B0, acc);
        acc = wmma_bf16(load_A_lds(A, LDH, 32, lane), B1, acc);
        store_tile_bf16(acc, kbf + m * 16 * LDKV + (ag - 1) * 32, LDKV, col, lane);
      }
    }
  }
  __syncthreads();

  // ---- stage 3: softmax attention (scalar) + build xc = [v_self|v_others] -
  for (int u = tid; u < MROWS * 32; u += 256) {
    const int r = u >> 5, f = u & 31;
    xc[r * LDXC + f] = vsb[r * LDQ + f];
  }
  if (tid < MROWS) {
    const int r = tid;
    const bf16* qr = qb + r * LDQ;
    float sq = ba[0];
#pragma unroll
    for (int f = 0; f < H8; ++f) sq += (float)qr[f] * Wa[f];
    float aw[7];
    float amax = -3.0e38f;
#pragma unroll
    for (int n = 0; n < 7; ++n) {
      const bf16* kr = kbf + r * LDKV + n * 32;
      float s = sq;
#pragma unroll
      for (int f = 0; f < H8; ++f) s += (float)kr[f] * Wa[H8 + f];
      s *= (1.0f / 16.0f);
      aw[n] = s;
      amax = fmaxf(amax, s);
    }
    float denom = 0.f;
#pragma unroll
    for (int n = 0; n < 7; ++n) { aw[n] = __expf(aw[n] - amax); denom += aw[n]; }
    const float inv = 1.0f / denom;
#pragma unroll 4
    for (int f = 0; f < H8; ++f) {
      float s = 0.f;
#pragma unroll
      for (int n = 0; n < 7; ++n) s += aw[n] * (float)vbf[r * LDKV + n * 32 + f];
      xc[r * LDXC + 32 + f] = (bf16)(s * inv);
    }
  }
  __syncthreads();

  // ---- stage 4: two MLP heads -> out[b], out[B + b] -----------------------
  mlp_branch(xc, t1, t2, W01, b01, W02, b02, W0, b0, out + b0row, tid);
  mlp_branch(xc, t1, t2, W11, b11, W12, b12, W1, b1, out + (size_t)Btot + b0row, tid);
}

// ---------------------------------------------------------------------------

extern "C" void kernel_launch(void* const* d_in, const int* in_sizes, int n_in,
                              void* d_out, int out_size, void* d_ws, size_t ws_size,
                              hipStream_t stream) {
  const float* obs = (const float*)d_in[0];
  const float* act = (const float*)d_in[1];
  const float* Ws  = (const float*)d_in[2];  const float* bs  = (const float*)d_in[3];
  const float* Wq  = (const float*)d_in[4];  const float* bq  = (const float*)d_in[5];
  const float* Wk  = (const float*)d_in[6];  const float* bk  = (const float*)d_in[7];
  const float* Wv  = (const float*)d_in[8];  const float* bv  = (const float*)d_in[9];
  const float* Wa  = (const float*)d_in[10]; const float* ba  = (const float*)d_in[11];
  const float* W01 = (const float*)d_in[12]; const float* b01 = (const float*)d_in[13];
  const float* W02 = (const float*)d_in[14]; const float* b02 = (const float*)d_in[15];
  const float* W0  = (const float*)d_in[16]; const float* b0  = (const float*)d_in[17];
  const float* W11 = (const float*)d_in[18]; const float* b11 = (const float*)d_in[19];
  const float* W12 = (const float*)d_in[20]; const float* b12 = (const float*)d_in[21];
  const float* W1  = (const float*)d_in[22]; const float* b1  = (const float*)d_in[23];

  const int Btot = in_sizes[0] / (NAG * PEROBS);   // 65536
  dim3 grid((Btot + MROWS - 1) / MROWS), block(256);
  const size_t ldsBytes = 143360;                  // 140 KB -> 2 blocks per WGP

  hipLaunchKernelGGL(critic_adq_kernel, grid, block, ldsBytes, stream,
                     obs, act, Ws, bs, Wq, bq, Wk, bk, Wv, bv, Wa, ba,
                     W01, b01, W02, b02, W0, b0, W11, b11, W12, b12, W1, b1,
                     (float*)d_out, Btot);
}
